// MultiHeadAttention_86870008529638
// MI455X (gfx1250) — compile-verified, matrix operand order checked by reading, not run on hardware
//
#include <hip/hip_runtime.h>
#include <hip/hip_bf16.h>

// ---------------------------------------------------------------------------
// Transformer-XL relative attention on MI455X (gfx1250, wave32, WMMA bf16)
// ---------------------------------------------------------------------------

typedef __attribute__((ext_vector_type(16))) __bf16 v16bf;
typedef __attribute__((ext_vector_type(8)))  __bf16 v8bf;
typedef __attribute__((ext_vector_type(8)))  float  v8f;

#define WMMA_BF16(a, b, c) \
  __builtin_amdgcn_wmma_f32_16x16x32_bf16(false, (a), false, (b), (short)0, (c), false, false)

// Load one wave-striped 16x32 bf16 operand (A or B role; layouts are identical
// for NT GEMMs with K contiguous). rowk points at &buf[row*ld + k0] where
// row = tile_row + (lane & 15). Per ISA 7.12.2:
//   lanes 0-15  : K = {0..7}  in v[0..7],  K = {16..23} in v[8..15]
//   lanes 16-31 : K = {8..15} in v[0..7],  K = {24..31} in v[8..15]
static __device__ inline v16bf ldAB(const __bf16* rowk, int lane) {
  int koff = (lane >> 4) << 3;
  v8bf lo = *reinterpret_cast<const v8bf*>(rowk + koff);
  v8bf hi = *reinterpret_cast<const v8bf*>(rowk + koff + 16);
  v16bf r;
#pragma unroll
  for (int i = 0; i < 8; ++i) { r[i] = lo[i]; r[8 + i] = hi[i]; }
  return r;
}

static __device__ inline v8f vzero8() {
  v8f z = {0.f, 0.f, 0.f, 0.f, 0.f, 0.f, 0.f, 0.f};
  return z;
}

// --------------------------- elementwise helpers ---------------------------

__global__ void cvt_f32_bf16_kernel(const float* __restrict__ src,
                                    __bf16* __restrict__ dst, int n) {
  int i = blockIdx.x * blockDim.x + threadIdx.x;
  if (i < n) dst[i] = (__bf16)src[i];
}

// r_b[m][d] = rel_emb[pos[m]][d], pos[m] = (L-1-m) + MAX_LEN = 4095 - m.
// Row m = 4095 is zero padding (never addressed by the shifted gather).
__global__ void build_r_kernel(const float* __restrict__ rel_emb,
                               __bf16* __restrict__ rb) {
  int i = blockIdx.x * blockDim.x + threadIdx.x;   // 4096*64 threads
  int m = i >> 6, d = i & 63;
  float v = (m < 4095) ? rel_emb[(4095 - m) * 64 + d] : 0.0f;
  rb[i] = (__bf16)v;
}

// vT[(h*64+d)][j] = vb[j][h*64+d]  (bf16, 1024 x 2048)
__global__ void transpose_v_kernel(const __bf16* __restrict__ vb,
                                   __bf16* __restrict__ vT) {
  int i = blockIdx.x * blockDim.x + threadIdx.x;   // 2048*1024 threads
  int j = i >> 10, d = i & 1023;
  vT[(long)d * 2048 + j] = vb[i];
}

// ------------------------------- WMMA GEMM ---------------------------------
// C[M,N] = A[M,K] * B[N,K]^T (+bias[n]).  One wave per 32x32 macro-tile
// (2x2 register blocking -> 4 WMMAs per 4 fragment loads per K-step),
// 8 waves per 256-thread block.  Epilogue modes:
//   outF : f32 store of acc+bias
//   out1 : bf16 store of acc+bias+add1   (add1 may be null)
//   out2 : bf16 store of acc+bias+add2
__global__ void gemm_bf16_nt_kernel(const __bf16* __restrict__ A, int lda,
                                    const __bf16* __restrict__ B, int ldb,
                                    int M, int N, int K,
                                    const float* __restrict__ bias,
                                    const float* __restrict__ add1,
                                    const float* __restrict__ add2,
                                    float* __restrict__ outF,
                                    __bf16* __restrict__ out1,
                                    __bf16* __restrict__ out2, int ldo) {
  int lane = threadIdx.x & 31;
  int wave = threadIdx.x >> 5;
  int tiles_n = N >> 5;                        // 32-wide macro tiles
  int tile = blockIdx.x * (blockDim.x >> 5) + wave;
  int tm = tile / tiles_n;
  int tn = tile - tm * tiles_n;
  if (tm >= (M >> 5)) return;                  // wave-uniform guard

  const __bf16* arow0 = A + (long)(tm * 32 + (lane & 15)) * lda;
  const __bf16* arow1 = arow0 + (long)16 * lda;
  const __bf16* brow0 = B + (long)(tn * 32 + (lane & 15)) * ldb;
  const __bf16* brow1 = brow0 + (long)16 * ldb;

  v8f c00 = vzero8(), c01 = vzero8(), c10 = vzero8(), c11 = vzero8();
  for (int k0 = 0; k0 < K; k0 += 32) {
    v16bf a0 = ldAB(arow0 + k0, lane);
    v16bf a1 = ldAB(arow1 + k0, lane);
    v16bf b0 = ldAB(brow0 + k0, lane);
    v16bf b1 = ldAB(brow1 + k0, lane);
    c00 = WMMA_BF16(a0, b0, c00);
    c01 = WMMA_BF16(a0, b1, c01);
    c10 = WMMA_BF16(a1, b0, c10);
    c11 = WMMA_BF16(a1, b1, c11);
  }

  auto store_tile = [&](const v8f& c, int mb, int nb) {
    int n = nb + (lane & 15);
    int mbase = mb + ((lane >> 4) << 3);
    float bs = bias ? bias[n] : 0.0f;
    float a1v = add1 ? add1[n] : 0.0f;
    float a2v = add2 ? add2[n] : 0.0f;
#pragma unroll
    for (int r = 0; r < 8; ++r) {
      long o = (long)(mbase + r) * ldo + n;
      float v = c[r] + bs;
      if (outF) outF[o] = v;
      if (out1) out1[o] = (__bf16)(v + a1v);
      if (out2) out2[o] = (__bf16)(v + a2v);
    }
  };
  store_tile(c00, tm * 32,      tn * 32);
  store_tile(c01, tm * 32,      tn * 32 + 16);
  store_tile(c10, tm * 32 + 16, tn * 32);
  store_tile(c11, tm * 32 + 16, tn * 32 + 16);
}

// --------------------------- flash attention (1 head) ----------------------
// One wave owns 16 query rows.  Per 32-key step:
//   S^T subtile = k_tile x qu^T  (WMMA, key-major => softmax state is
//   per-lane: query = lane&15, replicated across the two half-waves).
//   BD gather   : BD[l, j-l+2047] added before scaling (rel-shift fused).
//   P.V         : out^T = V^T x P^T; the two S^T C-tiles concatenated are
//   bit-exactly the B-matrix K-layout, so P never crosses lanes.
__global__ void attn_head_kernel(const __bf16* __restrict__ qu,  // (2048,1024) q+u_bias
                                 const __bf16* __restrict__ kb,  // (2048,1024)
                                 const __bf16* __restrict__ vT,  // (1024,2048)
                                 const __bf16* __restrict__ BD,  // (2048,4096)
                                 __bf16* __restrict__ ctx,       // (2048,1024)
                                 int head) {
  const int L = 2048, DM = 1024;
  int lane = threadIdx.x & 31;
  int wave = threadIdx.x >> 5;
  int col  = lane & 15;                 // query within the 16-row block
  int hi   = lane >> 4;
  int l0 = (blockIdx.x * (blockDim.x >> 5) + wave) << 4;
  int l  = l0 + col;

  // B operand (qu^T) is loop-invariant: two 32-wide d chunks.
  const __bf16* qrow = qu + (long)l * DM + head * 64;
  v16bf bq0 = ldAB(qrow, lane);
  v16bf bq1 = ldAB(qrow + 32, lane);

  v8f acc[4];
#pragma unroll
  for (int t = 0; t < 4; ++t) acc[t] = vzero8();

  float mrow = -1e30f, lrow = 0.0f;

  for (int j0 = 0; j0 < L; j0 += 32) {
    v8f s[2];
#pragma unroll
    for (int si = 0; si < 2; ++si) {
      const __bf16* krow = kb + (long)(j0 + 16 * si + col) * DM + head * 64;
      v16bf a0 = ldAB(krow, lane);
      v16bf a1 = ldAB(krow + 32, lane);
      v8f z = vzero8();
      z = WMMA_BF16(a0, bq0, z);
      z = WMMA_BF16(a1, bq1, z);
      s[si] = z;
    }

    // scores = (AC + BD_shifted) / sqrt(64); online softmax
    float p[16];
    float curmax = -1e30f;
#pragma unroll
    for (int si = 0; si < 2; ++si) {
      int jb = j0 + 16 * si + 8 * hi;           // key of register r=0
      const __bf16* bd = BD + (long)l * 4096 + (jb - l + 2047);
#pragma unroll
      for (int r = 0; r < 8; ++r) {
        float v = (s[si][r] + (float)bd[r]) * 0.125f;
        p[si * 8 + r] = v;
        curmax = fmaxf(curmax, v);
      }
    }
    curmax = fmaxf(curmax, __shfl_xor(curmax, 16, 32));
    float newm  = fmaxf(mrow, curmax);
    float alpha = __expf(mrow - newm);
    mrow = newm;

    float psum = 0.0f;
    v16bf pb;
#pragma unroll
    for (int i = 0; i < 16; ++i) {
      float e = __expf(p[i] - newm);
      psum += e;
      pb[i] = (__bf16)e;                        // lands in B-matrix K-layout
    }
    psum += __shfl_xor(psum, 16, 32);
    lrow = lrow * alpha + psum;

#pragma unroll
    for (int t = 0; t < 4; ++t) {
#pragma unroll
      for (int r = 0; r < 8; ++r) acc[t][r] *= alpha;
      const __bf16* vrow = vT + (long)(head * 64 + t * 16 + col) * 2048 + j0;
      v16bf va = ldAB(vrow, lane);
      acc[t] = WMMA_BF16(va, pb, acc[t]);       // out^T += V^T * P^T
    }
  }

  float inv = 1.0f / lrow;
#pragma unroll
  for (int t = 0; t < 4; ++t) {
    int dbase = head * 64 + t * 16 + 8 * hi;
#pragma unroll
    for (int r = 0; r < 8; ++r)
      ctx[(long)l * DM + dbase + r] = (__bf16)(acc[t][r] * inv);
  }
}

// --------------------------------- driver ----------------------------------

extern "C" void kernel_launch(void* const* d_in, const int* in_sizes, int n_in,
                              void* d_out, int out_size, void* d_ws, size_t ws_size,
                              hipStream_t stream) {
  const int L = 2048, DM = 1024, NH = 16;
  const float* Q   = (const float*)d_in[0];
  const float* K   = (const float*)d_in[1];
  const float* V   = (const float*)d_in[2];
  const float* Wq  = (const float*)d_in[3];
  const float* bq  = (const float*)d_in[4];
  const float* Wk  = (const float*)d_in[5];
  const float* bk  = (const float*)d_in[6];
  const float* Wv  = (const float*)d_in[7];
  const float* bv  = (const float*)d_in[8];
  const float* Wo  = (const float*)d_in[9];
  const float* bo  = (const float*)d_in[10];
  const float* rel = (const float*)d_in[11];
  const float* ub  = (const float*)d_in[12];  // (16,64) flat == column index
  const float* vb2 = (const float*)d_in[13];
  float* out = (float*)d_out;

  // workspace carve-up (bytes)
  char* ws = (char*)d_ws;
  const size_t SZ_LD  = (size_t)L * DM * 2;     // 4 MiB bf16 (L x 1024)
  const size_t SZ_W   = (size_t)DM * DM * 2;    // 2 MiB bf16 (1024 x 1024)
  __bf16* Qb   = (__bf16*)(ws);                    ws += SZ_LD;
  __bf16* Kb   = (__bf16*)(ws);                    ws += SZ_LD;
  __bf16* Vb   = (__bf16*)(ws);                    ws += SZ_LD;
  __bf16* Wqb  = (__bf16*)(ws);                    ws += SZ_W;
  __bf16* Wkb  = (__bf16*)(ws);                    ws += SZ_W;
  __bf16* Wvb  = (__bf16*)(ws);                    ws += SZ_W;
  __bf16* Wob  = (__bf16*)(ws);                    ws += SZ_W;
  __bf16* qub  = (__bf16*)(ws);                    ws += SZ_LD;
  __bf16* qvb  = (__bf16*)(ws);                    ws += SZ_LD;
  __bf16* kbuf = (__bf16*)(ws);                    ws += SZ_LD;
  __bf16* vbuf = (__bf16*)(ws);                    ws += SZ_LD;
  __bf16* vT   = (__bf16*)(ws);                    ws += SZ_LD;
  __bf16* rb   = (__bf16*)(ws);                    ws += (size_t)4096 * 64 * 2;
  __bf16* BD   = (__bf16*)(ws);                    ws += (size_t)L * 4096 * 2;
  __bf16* ctx  = (__bf16*)(ws);                    ws += SZ_LD;
  (void)ws_size; (void)in_sizes; (void)n_in; (void)out_size;

  const int T = 256;
  // 1) down-convert activations + weights to bf16
  cvt_f32_bf16_kernel<<<(L * DM + T - 1) / T, T, 0, stream>>>(Q, Qb, L * DM);
  cvt_f32_bf16_kernel<<<(L * DM + T - 1) / T, T, 0, stream>>>(K, Kb, L * DM);
  cvt_f32_bf16_kernel<<<(L * DM + T - 1) / T, T, 0, stream>>>(V, Vb, L * DM);
  cvt_f32_bf16_kernel<<<(DM * DM + T - 1) / T, T, 0, stream>>>(Wq, Wqb, DM * DM);
  cvt_f32_bf16_kernel<<<(DM * DM + T - 1) / T, T, 0, stream>>>(Wk, Wkb, DM * DM);
  cvt_f32_bf16_kernel<<<(DM * DM + T - 1) / T, T, 0, stream>>>(Wv, Wvb, DM * DM);
  cvt_f32_bf16_kernel<<<(DM * DM + T - 1) / T, T, 0, stream>>>(Wo, Wob, DM * DM);
  build_r_kernel<<<(4096 * 64) / T, T, 0, stream>>>(rel, rb);

  // 2) projections: M=2048, N=1024 -> (64*32) 32x32 tiles / 8 waves = 256 blocks
  gemm_bf16_nt_kernel<<<256, 256, 0, stream>>>(Qb, DM, Wqb, DM, L, DM, DM,
      bq, ub, vb2, nullptr, qub, qvb, DM);            // qu = q+u, qv = q+v
  gemm_bf16_nt_kernel<<<256, 256, 0, stream>>>(Kb, DM, Wkb, DM, L, DM, DM,
      bk, nullptr, nullptr, nullptr, kbuf, nullptr, DM);
  gemm_bf16_nt_kernel<<<256, 256, 0, stream>>>(Vb, DM, Wvb, DM, L, DM, DM,
      bv, nullptr, nullptr, nullptr, vbuf, nullptr, DM);
  transpose_v_kernel<<<(L * DM) / T, T, 0, stream>>>(vbuf, vT);

  // 3) per head: BD = (q+v) @ r^T  (M=2048,N=4096,K=64), then flash attention.
  //    Serialized on the stream so the single 16 MiB BD buffer is reused.
  for (int h = 0; h < NH; ++h) {
    gemm_bf16_nt_kernel<<<1024, 256, 0, stream>>>(qvb + h * 64, DM, rb, 64,
        L, 4096, 64, nullptr, nullptr, nullptr, nullptr, BD, nullptr, 4096);
    attn_head_kernel<<<16, 256, 0, stream>>>(qub, kbuf, vT, BD, ctx, h);
  }

  // 4) output projection -> f32 d_out
  gemm_bf16_nt_kernel<<<256, 256, 0, stream>>>(ctx, DM, Wob, DM, L, DM, DM,
      bo, nullptr, nullptr, out, nullptr, nullptr, DM);
}
